// NN_42502996361823
// MI455X (gfx1250) — matrix operand order repaired and checked by reading, not compiled, not run on hardware
//
#include <hip/hip_runtime.h>
#include <hip/hip_bf16.h>
#include <stdint.h>

#define CN     64
#define NNODE  16384
#define NEDGE  32768
#define NBATCH 512
#define NPASS  4

typedef __attribute__((ext_vector_type(16))) __bf16   v16bf;
typedef __attribute__((ext_vector_type(8)))  float    v8f;
typedef __attribute__((ext_vector_type(4)))  uint32_t v4u;

union Frag16 { v4u u[2]; v16bf v; };

__device__ __forceinline__ float sigf(float v) { return 1.0f / (1.0f + __expf(-v)); }

// ---------------- prep kernels ----------------

// out = relu(x @ lin0_w.T + lin0_b)   x:[N,28] w:[64,28]
__global__ __launch_bounds__(256) void k_lin0(const float* __restrict__ x,
                                              const float* __restrict__ w,
                                              const float* __restrict__ b,
                                              float* __restrict__ out) {
  int idx = blockIdx.x * 256 + threadIdx.x;      // n*64 + c
  int n = idx >> 6, c = idx & 63;
  float acc = b[c];
#pragma unroll
  for (int j = 0; j < 28; ++j) acc += x[n * 28 + j] * w[c * 28 + j];
  out[idx] = fmaxf(acc, 0.0f);
}

// hid = bf16(relu(edge_attr @ nn1_w.T + nn1_b))   [E,128]
__global__ __launch_bounds__(256) void k_hid(const float* __restrict__ ea,
                                             const float* __restrict__ w,
                                             const float* __restrict__ b,
                                             __bf16* __restrict__ hid) {
  int idx = blockIdx.x * 256 + threadIdx.x;      // e*128 + k
  int e = idx >> 7, k = idx & 127;
  float acc = b[k];
#pragma unroll
  for (int j = 0; j < 4; ++j) acc += ea[e * 4 + j] * w[k * 4 + j];
  hid[idx] = (__bf16)fmaxf(acc, 0.0f);
}

// nn2_w f32[4096,128] -> bf16 same layout
__global__ __launch_bounds__(256) void k_cvtw(const float* __restrict__ w,
                                              __bf16* __restrict__ wb) {
  int idx = blockIdx.x * 256 + threadIdx.x;
  wb[idx] = (__bf16)w[idx];
}

// nn2bT[o*64+i] = bf16(nn2_b[i*64+o])   (transposed bias for B-fragment layout)
__global__ __launch_bounds__(256) void k_cvtb(const float* __restrict__ b,
                                              __bf16* __restrict__ bt) {
  int idx = blockIdx.x * 256 + threadIdx.x;      // o*64 + i
  int o = idx >> 6, i = idx & 63;
  bt[idx] = (__bf16)b[i * 64 + o];
}

// deg[dst[e]] += 1
__global__ __launch_bounds__(256) void k_deg(const int* __restrict__ dst,
                                             float* __restrict__ deg) {
  int e = blockIdx.x * 256 + threadIdx.x;
  if (e < NEDGE) atomicAdd(&deg[dst[e]], 1.0f);
}

// ---------------- fused NNConv edge GEMM (WMMA bf16, double-buffered) ----------------
// msg[e,o] = sum_{i,k} x_src[e,i]*hid[e,k]*nn2_w[i*64+o,k] + sum_i x_src[e,i]*nn2_b[i*64+o]
// 1 wave = 16 edges; 8 waves/block. B fragments ping-pong between two named register
// sets so 8 b128 loads stay in flight behind 4 chained WMMAs (no loadcnt-0 drains).

#define WMMA_BF16(a, bmat, c) \
  __builtin_amdgcn_wmma_f32_16x16x32_bf16(false, (a), false, (bmat), (short)0, (c), false, false)

// load B fragments for all 4 N-tiles of flat step (i*4+ks) into Bn##0..3
#define LOADB(Bn, stepExpr) do {                                                   \
    const int st_ = (stepExpr);                                                    \
    const __bf16* p_ = bbase + (size_t)((st_ >> 2) * 64) * 128 + (st_ & 3) * 32;   \
    (Bn##0).u[0] = *(const v4u*)(p_);             (Bn##0).u[1] = *(const v4u*)(p_ + 8);            \
    (Bn##1).u[0] = *(const v4u*)(p_ + 16 * 128);  (Bn##1).u[1] = *(const v4u*)(p_ + 16 * 128 + 8); \
    (Bn##2).u[0] = *(const v4u*)(p_ + 32 * 128);  (Bn##2).u[1] = *(const v4u*)(p_ + 32 * 128 + 8); \
    (Bn##3).u[0] = *(const v4u*)(p_ + 48 * 128);  (Bn##3).u[1] = *(const v4u*)(p_ + 48 * 128 + 8); \
  } while (0)

// one (i,ks) step: prefetch next B set, scale cached A frag by x_i, 4 WMMAs
#define WSTEP(Bc, Bn, i_, ks_, pf_) do {                                           \
    if (pf_) LOADB(Bn, (i_) * 4 + (ks_) + 1);                                      \
    Frag16 As_;                                                                    \
    _Pragma("unroll")                                                              \
    for (int t = 0; t < 16; ++t) As_.v[t] = (__bf16)(xv * (float)Afr[ks_].v[t]);   \
    acc0 = WMMA_BF16(As_.v, (Bc##0).v, acc0);                                      \
    acc1 = WMMA_BF16(As_.v, (Bc##1).v, acc1);                                      \
    acc2 = WMMA_BF16(As_.v, (Bc##2).v, acc2);                                      \
    acc3 = WMMA_BF16(As_.v, (Bc##3).v, acc3);                                      \
  } while (0)

// bias chunk cc: A built from x itself, B = nn2_b^T
#define BIAS_STEP(cc_) do {                                                        \
    Frag16 As_;                                                                    \
    _Pragma("unroll")                                                              \
    for (int t = 0; t < 16; ++t) {                                                 \
      int kl_ = hh * 16 + t;                                                       \
      As_.v[t] = (__bf16)xs[wave][mrow][(cc_) * 32 + kl_];                         \
    }                                                                              \
    const __bf16* bp_ = nn2bT + (size_t)mrow * 64 + hh * 16 + (cc_) * 32;          \
    Frag16 Bt0, Bt1, Bt2, Bt3;                                                     \
    Bt0.u[0] = *(const v4u*)(bp_);            Bt0.u[1] = *(const v4u*)(bp_ + 8);             \
    Bt1.u[0] = *(const v4u*)(bp_ + 16 * 64);  Bt1.u[1] = *(const v4u*)(bp_ + 16 * 64 + 8);   \
    Bt2.u[0] = *(const v4u*)(bp_ + 32 * 64);  Bt2.u[1] = *(const v4u*)(bp_ + 32 * 64 + 8);   \
    Bt3.u[0] = *(const v4u*)(bp_ + 48 * 64);  Bt3.u[1] = *(const v4u*)(bp_ + 48 * 64 + 8);   \
    acc0 = WMMA_BF16(As_.v, Bt0.v, acc0);                                          \
    acc1 = WMMA_BF16(As_.v, Bt1.v, acc1);                                          \
    acc2 = WMMA_BF16(As_.v, Bt2.v, acc2);                                          \
    acc3 = WMMA_BF16(As_.v, Bt3.v, acc3);                                          \
  } while (0)

// scatter one N-tile: C/D layout row M = v (lanes 0-15) / v+8 (lanes 16-31), col q*16+lane%16
#define SCATTER(q_, accv) do {                                                     \
    _Pragma("unroll")                                                              \
    for (int v = 0; v < 8; ++v) {                                                  \
      int row_ = hh ? (v + 8) : v;                                                 \
      atomicAdd(&agg[(size_t)dsts[wave][row_] * 64 + (q_) * 16 + mrow], accv[v]);  \
    }                                                                              \
  } while (0)

__global__ __launch_bounds__(256) void k_edge(const __bf16* __restrict__ hid,     // [E,128]
                                              const __bf16* __restrict__ nn2wb,   // [4096,128]
                                              const __bf16* __restrict__ nn2bT,   // [64,64] (o-major)
                                              const float* __restrict__ out,      // [N,64]
                                              const int* __restrict__ eidx,       // [2,E]
                                              float* __restrict__ agg) {          // [N,64]
  __shared__ float xs[8][16][64];
  __shared__ int   dsts[8][16];

  const int wave = threadIdx.x >> 5;
  const int lane = threadIdx.x & 31;
  const int tile = blockIdx.x * 8 + wave;
  const int e0   = tile * 16;
  const int mrow = lane & 15;
  const int hh   = (lane < 16) ? 0 : 1;   // lane half

  const int* __restrict__ srcp = eidx;
  const int* __restrict__ dstp = eidx + NEDGE;

  if (lane < 16) dsts[wave][lane] = dstp[e0 + lane];
#pragma unroll 4
  for (int r = 0; r < 16; ++r) {
    int s = srcp[e0 + r];
    float2 v = *(const float2*)(out + (size_t)s * 64 + lane * 2);
    *(float2*)&xs[wave][r][lane * 2] = v;
  }
  __syncthreads();

  // A fragments: hid row of this lane's edge, 4 K-chunks of 32 (bf16 A layout)
  const __bf16* hrow = hid + (size_t)(e0 + mrow) * 128;
  Frag16 Afr[4];
#pragma unroll
  for (int ks = 0; ks < 4; ++ks) {
    int kb = ks * 32 + hh * 16;
    Afr[ks].u[0] = *(const v4u*)(hrow + kb);
    Afr[ks].u[1] = *(const v4u*)(hrow + kb + 8);
  }

  const __bf16* bbase = nn2wb + (size_t)mrow * 128 + hh * 16;

  v8f acc0 = {0.f, 0.f, 0.f, 0.f, 0.f, 0.f, 0.f, 0.f};
  v8f acc1 = acc0, acc2 = acc0, acc3 = acc0;

  Frag16 Ba0, Ba1, Ba2, Ba3, Bb0, Bb1, Bb2, Bb3;
  LOADB(Ba, 0);

  for (int i = 0; i < 64; ++i) {
    const float xv = xs[wave][mrow][i];          // x_src[row, i] (LDS broadcast)
    WSTEP(Ba, Bb, i, 0, 1);
    WSTEP(Bb, Ba, i, 1, 1);
    WSTEP(Ba, Bb, i, 2, 1);
    WSTEP(Bb, Ba, i, 3, (i < 63));
  }

  BIAS_STEP(0);
  BIAS_STEP(1);

  SCATTER(0, acc0);
  SCATTER(1, acc1);
  SCATTER(2, acc2);
  SCATTER(3, acc3);
}

// ---------------- node update: scatter-mean + root + GRU ----------------
__global__ __launch_bounds__(64) void k_node(const float* __restrict__ agg,
                                             const float* __restrict__ deg,
                                             const float* __restrict__ root_w,   // [64,64]
                                             const float* __restrict__ conv_b,
                                             const float* __restrict__ wih,      // [192,64]
                                             const float* __restrict__ whh,      // [192,64]
                                             const float* __restrict__ bih,
                                             const float* __restrict__ bhh,
                                             float* __restrict__ out) {          // in/out [N,64]
  __shared__ float orow[64];
  __shared__ float mrow[64];
  const int n = blockIdx.x, c = threadIdx.x;
  orow[c] = out[(size_t)n * 64 + c];
  __syncthreads();

  float cnt = fmaxf(deg[n], 1.0f);
  float acc = agg[(size_t)n * 64 + c] / cnt + conv_b[c];
#pragma unroll 8
  for (int i = 0; i < 64; ++i) acc += orow[i] * root_w[i * 64 + c];
  mrow[c] = fmaxf(acc, 0.0f);
  __syncthreads();

  float ir = bih[c], iz = bih[64 + c], in_ = bih[128 + c];
  float hr = bhh[c], hz = bhh[64 + c], hn  = bhh[128 + c];
#pragma unroll 4
  for (int j = 0; j < 64; ++j) {
    float mj = mrow[j], hj = orow[j];
    ir  += mj * wih[c * 64 + j];
    iz  += mj * wih[(64 + c) * 64 + j];
    in_ += mj * wih[(128 + c) * 64 + j];
    hr  += hj * whh[c * 64 + j];
    hz  += hj * whh[(64 + c) * 64 + j];
    hn  += hj * whh[(128 + c) * 64 + j];
  }
  float r = sigf(ir + hr);
  float z = sigf(iz + hz);
  float nn = tanhf(in_ + r * hn);
  out[(size_t)n * 64 + c] = (1.0f - z) * nn + z * orow[c];
}

// ---------------- Set2Set step (1 block per graph, 32 nodes/graph) ----------------
__global__ __launch_bounds__(64) void k_s2s(const float* __restrict__ out,
                                            const float* __restrict__ wih,   // [256,128]
                                            const float* __restrict__ whh,   // [256,64]
                                            const float* __restrict__ bih,
                                            const float* __restrict__ bhh,
                                            float* __restrict__ qstar,       // [B,128]
                                            float* __restrict__ hl,          // [B,64]
                                            float* __restrict__ cl) {        // [B,64]
  __shared__ float qs[128];
  __shared__ float hls[64];
  __shared__ float qv[64];
  __shared__ float ev[32];
  __shared__ float av[32];
  const int b = blockIdx.x, c = threadIdx.x;

  qs[c] = qstar[(size_t)b * 128 + c];
  qs[64 + c] = qstar[(size_t)b * 128 + 64 + c];
  hls[c] = hl[(size_t)b * 64 + c];
  __syncthreads();

  float gI = bih[c] + bhh[c];
  float gF = bih[64 + c] + bhh[64 + c];
  float gG = bih[128 + c] + bhh[128 + c];
  float gO = bih[192 + c] + bhh[192 + c];
  for (int j = 0; j < 128; ++j) {
    float q = qs[j];
    gI += q * wih[c * 128 + j];
    gF += q * wih[(64 + c) * 128 + j];
    gG += q * wih[(128 + c) * 128 + j];
    gO += q * wih[(192 + c) * 128 + j];
  }
  for (int j = 0; j < 64; ++j) {
    float hj = hls[j];
    gI += hj * whh[c * 64 + j];
    gF += hj * whh[(64 + c) * 64 + j];
    gG += hj * whh[(128 + c) * 64 + j];
    gO += hj * whh[(192 + c) * 64 + j];
  }
  float cn = sigf(gF) * cl[(size_t)b * 64 + c] + sigf(gI) * tanhf(gG);
  float hn = sigf(gO) * tanhf(cn);
  cl[(size_t)b * 64 + c] = cn;
  hl[(size_t)b * 64 + c] = hn;
  qv[c] = hn;
  __syncthreads();

  // attention over this graph's 32 nodes
  if (c < 32) {
    int n = b * 32 + c;
    float e = 0.0f;
    for (int j = 0; j < 64; ++j) e += out[(size_t)n * 64 + j] * qv[j];
    ev[c] = e;
  }
  __syncthreads();
  float mx = ev[0];
  for (int j = 1; j < 32; ++j) mx = fmaxf(mx, ev[j]);
  if (c < 32) av[c] = __expf(ev[c] - mx);
  __syncthreads();
  float s = 0.0f;
  for (int j = 0; j < 32; ++j) s += av[j];
  float r = 0.0f;
  for (int j = 0; j < 32; ++j) r += av[j] * out[(size_t)(b * 32 + j) * 64 + c];
  r /= s;
  qstar[(size_t)b * 128 + c] = qv[c];
  qstar[(size_t)b * 128 + 64 + c] = r;
}

// ---------------- output heads ----------------
__global__ __launch_bounds__(64) void k_fp(const float* __restrict__ qstar,
                                           const float* __restrict__ w,   // [64,128]
                                           const float* __restrict__ b,
                                           float* __restrict__ dout) {    // fp at dout+512
  __shared__ float qs[128];
  const int g = blockIdx.x, c = threadIdx.x;
  qs[c] = qstar[(size_t)g * 128 + c];
  qs[64 + c] = qstar[(size_t)g * 128 + 64 + c];
  __syncthreads();
  float acc = b[c];
  for (int j = 0; j < 128; ++j) acc += qs[j] * w[c * 128 + j];
  dout[512 + (size_t)g * 64 + c] = fmaxf(acc, 0.0f);
}

__global__ __launch_bounds__(64) void k_y(const float* __restrict__ w,    // [1,64]
                                          const float* __restrict__ b,
                                          float* __restrict__ dout) {
  const int g = blockIdx.x * 64 + threadIdx.x;
  const float* fp = dout + 512 + (size_t)g * 64;
  float acc = b[0];
  for (int j = 0; j < 64; ++j) acc += fp[j] * w[j];
  dout[g] = acc;
}

// ---------------- launcher ----------------
extern "C" void kernel_launch(void* const* d_in, const int* in_sizes, int n_in,
                              void* d_out, int out_size, void* d_ws, size_t ws_size,
                              hipStream_t stream) {
  (void)in_sizes; (void)n_in; (void)out_size; (void)ws_size;
  const float* x        = (const float*)d_in[0];
  const float* ea       = (const float*)d_in[1];
  const float* lin0_w   = (const float*)d_in[2];
  const float* lin0_b   = (const float*)d_in[3];
  const float* nn1_w    = (const float*)d_in[4];
  const float* nn1_b    = (const float*)d_in[5];
  const float* nn2_w    = (const float*)d_in[6];
  const float* nn2_b    = (const float*)d_in[7];
  const float* root_w   = (const float*)d_in[8];
  const float* conv_b   = (const float*)d_in[9];
  const float* gru_wih  = (const float*)d_in[10];
  const float* gru_whh  = (const float*)d_in[11];
  const float* gru_bih  = (const float*)d_in[12];
  const float* gru_bhh  = (const float*)d_in[13];
  const float* lstm_wih = (const float*)d_in[14];
  const float* lstm_whh = (const float*)d_in[15];
  const float* lstm_bih = (const float*)d_in[16];
  const float* lstm_bhh = (const float*)d_in[17];
  const float* lin1_w   = (const float*)d_in[18];
  const float* lin1_b   = (const float*)d_in[19];
  const float* lin2_w   = (const float*)d_in[20];
  const float* lin2_b   = (const float*)d_in[21];
  const int*   eidx     = (const int*)d_in[22];
  float* dout = (float*)d_out;

  // workspace carve-up
  char* w = (char*)d_ws;
  auto align256 = [](size_t v) { return (v + 255) & ~(size_t)255; };
  size_t off = 0;
  float* outbuf = (float*)(w + off); off = align256(off + (size_t)NNODE * 64 * 4);
  float* agg    = (float*)(w + off); off = align256(off + (size_t)NNODE * 64 * 4);
  float* deg    = (float*)(w + off); off = align256(off + (size_t)NNODE * 4);
  __bf16* hid   = (__bf16*)(w + off); off = align256(off + (size_t)NEDGE * 128 * 2);
  __bf16* nn2wb = (__bf16*)(w + off); off = align256(off + (size_t)4096 * 128 * 2);
  __bf16* nn2bT = (__bf16*)(w + off); off = align256(off + (size_t)4096 * 2);
  float* qstar  = (float*)(w + off); off = align256(off + (size_t)NBATCH * 128 * 4);
  float* hl     = (float*)(w + off); off = align256(off + (size_t)NBATCH * 64 * 4);
  float* cl     = (float*)(w + off); off = align256(off + (size_t)NBATCH * 64 * 4);

  // ---- pass-invariant prep ----
  hipMemsetAsync(deg, 0, (size_t)NNODE * 4, stream);
  k_deg <<<NEDGE / 256, 256, 0, stream>>>(eidx + NEDGE, deg);
  k_lin0<<<NNODE * 64 / 256, 256, 0, stream>>>(x, lin0_w, lin0_b, outbuf);
  k_hid <<<NEDGE * 128 / 256, 256, 0, stream>>>(ea, nn1_w, nn1_b, hid);
  k_cvtw<<<4096 * 128 / 256, 256, 0, stream>>>(nn2_w, nn2wb);
  k_cvtb<<<4096 / 256, 256, 0, stream>>>(nn2_b, nn2bT);

  // ---- message passing ----
  for (int p = 0; p < NPASS; ++p) {
    hipMemsetAsync(agg, 0, (size_t)NNODE * 64 * 4, stream);
    k_edge<<<NEDGE / 128, 256, 0, stream>>>(hid, nn2wb, nn2bT, outbuf, eidx, agg);
    k_node<<<NNODE, 64, 0, stream>>>(agg, deg, root_w, conv_b,
                                     gru_wih, gru_whh, gru_bih, gru_bhh, outbuf);
  }

  // ---- Set2Set ----
  hipMemsetAsync(qstar, 0, (size_t)NBATCH * 128 * 4, stream);
  hipMemsetAsync(hl, 0, (size_t)NBATCH * 64 * 4, stream);
  hipMemsetAsync(cl, 0, (size_t)NBATCH * 64 * 4, stream);
  for (int p = 0; p < NPASS; ++p)
    k_s2s<<<NBATCH, 64, 0, stream>>>(outbuf, lstm_wih, lstm_whh, lstm_bih, lstm_bhh,
                                     qstar, hl, cl);

  // ---- heads: dout = [y(512), fp(512*64)] ----
  k_fp<<<NBATCH, 64, 0, stream>>>(qstar, lin1_w, lin1_b, dout);
  k_y <<<NBATCH / 64, 64, 0, stream>>>(lin2_w, lin2_b, dout);
}
